// minCUTPooling_6777458393290
// MI455X (gfx1250) — compile-verified, matrix operand order checked by reading, not run on hardware
//
#include <hip/hip_runtime.h>
#include <hip/hip_bf16.h>
#include <stdint.h>

// GAT (2 layers, 4 heads x 64ch, head-mean) + 2 FC/BN heads for gfx1250.
// GEMMs use v_wmma_f32_16x16x32_bf16 (wave32 WMMA), edge softmax-aggregation
// uses L2-resident fp32 global atomics (xl is 102MB, fits the 192MB L2).

typedef __attribute__((ext_vector_type(16))) __bf16 v16bf;
typedef __attribute__((ext_vector_type(8)))  __bf16 v8bf;
typedef __attribute__((ext_vector_type(8)))  float  v8f;

#define NEG_SLOPE 0.2f
#define BN_EPS    1e-5f

// Monotone order-preserving float<->uint encoding for atomicMax on floats
// (handles negatives from leaky-relu).  max(enc(a),enc(b)) == enc(max(a,b)).
static __device__ __forceinline__ unsigned ordEncode(float f) {
  unsigned b = __float_as_uint(f);
  return (b & 0x80000000u) ? ~b : (b | 0x80000000u);
}
static __device__ __forceinline__ float ordDecode(unsigned e) {
  unsigned b = (e & 0x80000000u) ? (e & 0x7fffffffu) : ~e;
  return __uint_as_float(b);
}
#define ORD_NEG_INF 0x007FFFFFu   // ordEncode(-inf)

// packed fp32x2 -> bf16x2 (one v_cvt_pk_bf16_f32 + b32 store)
union BfPack { unsigned u; __bf16 h[2]; };
static __device__ __forceinline__ unsigned packBf2(float a, float b) {
  BfPack p; p.h[0] = (__bf16)a; p.h[1] = (__bf16)b; return p.u;
}

// ---------------------------------------------------------------- fill util
__global__ __launch_bounds__(256) void fill_u32(unsigned* p, unsigned v, long long n) {
  long long i = (long long)blockIdx.x * blockDim.x + threadIdx.x;
  if (i < n) p[i] = v;
}

// ------------------------------------------------------------ WMMA GEMM
// out[N,256] = A[N,DIN] @ W[DIN,256]   (fp32 in, bf16 WMMA, fp32 accumulate)
// Block: 256 threads = 8 waves, covers 32 rows x 256 cols.
// Wave w owns col tiles {2w, 2w+1} x row tiles {0,1}  -> 4 wmma per K chunk.
// sA is row-major [row][k]; sWt is TRANSPOSED [n][k] so that each lane's A and
// B fragments are two contiguous 16B runs -> two ds_load_b128, no repacking.
template <int DIN>
__global__ __launch_bounds__(256)
void gat_gemm_wmma(const float* __restrict__ A, const float* __restrict__ W,
                   float* __restrict__ out, int nrows) {
  constexpr int KCH = DIN / 32;
  __shared__ __bf16 sA[32 * 32];     // [row 0..31][k 0..31]
  __shared__ __bf16 sWt[256 * 32];   // [n 0..255][k 0..31]
  unsigned* sA32 = (unsigned*)sA;

  const int tid  = threadIdx.x;
  const int lane = tid & 31;
  const int wave = tid >> 5;
  const int rowBase = blockIdx.x * 32;
  const int half = lane >> 4;   // wave32 lane-half (K-group select, ISA 7.12.2)
  const int mloc = lane & 15;

  v8f cAcc[4] = {v8f{}, v8f{}, v8f{}, v8f{}};

  for (int kc = 0; kc < KCH; ++kc) {
    __syncthreads();
    // Stage A chunk (32 rows x 32 k), branchless: clamp row (clamped rows only
    // feed output rows that are never stored), float2 loads, packed bf16 stores.
    #pragma unroll
    for (int i = 0; i < 2; ++i) {
      int e2 = i * 256 + tid;              // pair index: 512 pairs total
      int r = e2 >> 4, kp = e2 & 15;
      int row = rowBase + r; row = row < nrows ? row : nrows - 1;
      float2 v = *(const float2*)(A + (size_t)row * DIN + kc * 32 + kp * 2);
      sA32[e2] = packBf2(v.x, v.y);
    }
    // Stage W chunk transposed: thread tid owns column n=tid; K pairs packed,
    // each lane writes its 64B sWt row contiguously.
    {
      unsigned pk[16];
      #pragma unroll
      for (int kp = 0; kp < 16; ++kp) {
        const float* wp = W + (size_t)(kc * 32 + 2 * kp) * 256 + tid;
        pk[kp] = packBf2(wp[0], wp[256]);
      }
      unsigned* dstp = (unsigned*)(sWt + tid * 32);
      #pragma unroll
      for (int kp = 0; kp < 16; ++kp) dstp[kp] = pk[kp];
    }
    __syncthreads();

    // A fragments (16x32 bf16): lane element order is K = half*8..+8 then
    // 16+half*8..+8 -> two contiguous v8bf (b128) loads.
    v16bf aF[2], bF[2];
    #pragma unroll
    for (int rt = 0; rt < 2; ++rt) {
      const v8bf* p = (const v8bf*)(sA + (rt * 16 + mloc) * 32 + half * 8);
      v8bf lo = p[0], hi = p[2];
      aF[rt] = __builtin_shufflevector(lo, hi, 0, 1, 2, 3, 4, 5, 6, 7,
                                       8, 9, 10, 11, 12, 13, 14, 15);
    }
    // B fragments (32x16 bf16): same K grouping along sWt rows.
    #pragma unroll
    for (int ct = 0; ct < 2; ++ct) {
      int n = (wave * 2 + ct) * 16 + mloc;
      const v8bf* p = (const v8bf*)(sWt + n * 32 + half * 8);
      v8bf lo = p[0], hi = p[2];
      bF[ct] = __builtin_shufflevector(lo, hi, 0, 1, 2, 3, 4, 5, 6, 7,
                                       8, 9, 10, 11, 12, 13, 14, 15);
    }
    #pragma unroll
    for (int rt = 0; rt < 2; ++rt)
      #pragma unroll
      for (int ct = 0; ct < 2; ++ct)
        cAcc[rt * 2 + ct] = __builtin_amdgcn_wmma_f32_16x16x32_bf16(
            false, aF[rt], false, bF[ct], (short)0, cAcc[rt * 2 + ct],
            false, false);
  }

  // Store C tiles: lane holds col (lane&15), rows r + 8*(lane>>4).
  // Fast path: whole 32-row block in range (always true when nrows%32==0).
  if (rowBase + 32 <= nrows) {
    #pragma unroll
    for (int rt = 0; rt < 2; ++rt) {
      #pragma unroll
      for (int ct = 0; ct < 2; ++ct) {
        int n = (wave * 2 + ct) * 16 + mloc;
        size_t base = (size_t)(rowBase + rt * 16 + half * 8) * 256 + n;
        #pragma unroll
        for (int r = 0; r < 8; ++r)
          out[base + (size_t)r * 256] = cAcc[rt * 2 + ct][r];
      }
    }
  } else {
    #pragma unroll
    for (int rt = 0; rt < 2; ++rt) {
      #pragma unroll
      for (int ct = 0; ct < 2; ++ct) {
        int n = (wave * 2 + ct) * 16 + mloc;
        #pragma unroll
        for (int r = 0; r < 8; ++r) {
          int m = rowBase + rt * 16 + r + half * 8;
          if (m < nrows) out[(size_t)m * 256 + n] = cAcc[rt * 2 + ct][r];
        }
      }
    }
  }
}

// ---------------------------------------------------- attention projections
// alpha_src[n,h] = sum_c xl[n,h,c]*a_src[h,c] ; likewise alpha_dst.
__global__ __launch_bounds__(256)
void alpha_kernel(const float* __restrict__ xl, const float* __restrict__ aS,
                  const float* __restrict__ aD, float* __restrict__ oS,
                  float* __restrict__ oD, int nnodes) {
  long long gid = (long long)blockIdx.x * blockDim.x + threadIdx.x;
  if (gid >= (long long)nnodes * 4) return;
  int n = (int)(gid >> 2), h = (int)(gid & 3);
  const float* row = xl + (size_t)n * 256 + h * 64;
  const float* as = aS + h * 64;
  const float* ad = aD + h * 64;
  float ss = 0.f, sd = 0.f;
  #pragma unroll 8
  for (int c = 0; c < 64; ++c) { float v = row[c]; ss += v * as[c]; sd += v * ad[c]; }
  oS[gid] = ss; oD[gid] = sd;
}

static __device__ __forceinline__ void edge_ids(long long idx, const long long* ei,
                                                int nedges, int& s, int& d) {
  if (idx < nedges) { s = (int)ei[idx]; d = (int)ei[nedges + idx]; }
  else { s = d = (int)(idx - nedges); }           // appended self-loops
}

// Pass A: segment max of leaky-relu'd logits per (dst, head)
__global__ __launch_bounds__(256)
void edge_max(const long long* __restrict__ ei, const float* __restrict__ aS,
              const float* __restrict__ aD, unsigned* __restrict__ mEnc,
              int nedges, int nnodes) {
  long long gid = (long long)blockIdx.x * blockDim.x + threadIdx.x;
  if (gid >= (long long)(nedges + nnodes) * 4) return;
  long long idx = gid >> 2; int h = (int)(gid & 3);
  int s, d; edge_ids(idx, ei, nedges, s, d);
  float ev = aS[(size_t)s * 4 + h] + aD[(size_t)d * 4 + h];
  ev = ev > 0.f ? ev : NEG_SLOPE * ev;
  atomicMax(&mEnc[(size_t)d * 4 + h], ordEncode(ev));
}

// Pass B: segment sum of exp(e - m[dst])
__global__ __launch_bounds__(256)
void edge_sum(const long long* __restrict__ ei, const float* __restrict__ aS,
              const float* __restrict__ aD, const unsigned* __restrict__ mEnc,
              float* __restrict__ denom, int nedges, int nnodes) {
  long long gid = (long long)blockIdx.x * blockDim.x + threadIdx.x;
  if (gid >= (long long)(nedges + nnodes) * 4) return;
  long long idx = gid >> 2; int h = (int)(gid & 3);
  int s, d; edge_ids(idx, ei, nedges, s, d);
  float ev = aS[(size_t)s * 4 + h] + aD[(size_t)d * 4 + h];
  ev = ev > 0.f ? ev : NEG_SLOPE * ev;
  float ex = __expf(ev - ordDecode(mEnc[(size_t)d * 4 + h]));
  atomicAdd(&denom[(size_t)d * 4 + h], ex);
}

// Pass C: one wave per edge; lane owns 8 of 256 channels (head = lane>>3).
// Gathers xl[src] (L2-resident), scatter-adds alpha-weighted row into acc[dst].
__global__ __launch_bounds__(256)
void edge_msg(const long long* __restrict__ ei, const float* __restrict__ aS,
              const float* __restrict__ aD, const unsigned* __restrict__ mEnc,
              const float* __restrict__ denom, const float* __restrict__ xl,
              float* __restrict__ acc, int nedges, int nnodes) {
  int lane = threadIdx.x & 31;
  long long eidx = (long long)blockIdx.x * 8 + (threadIdx.x >> 5);
  if (eidx >= (long long)(nedges + nnodes)) return;   // wave-uniform exit
  int s, d; edge_ids(eidx, ei, nedges, s, d);
  int h = lane >> 3;
  float ev = aS[(size_t)s * 4 + h] + aD[(size_t)d * 4 + h];
  ev = ev > 0.f ? ev : NEG_SLOPE * ev;
  float ex = __expf(ev - ordDecode(mEnc[(size_t)d * 4 + h]));
  float alpha = ex / (denom[(size_t)d * 4 + h] + 1e-16f);
  const float* xrow = xl + (size_t)s * 256 + lane * 8;
  float* arow = acc + (size_t)d * 256 + lane * 8;
  float4 v0 = *(const float4*)(xrow);
  float4 v1 = *(const float4*)(xrow + 4);
  atomicAdd(arow + 0, v0.x * alpha); atomicAdd(arow + 1, v0.y * alpha);
  atomicAdd(arow + 2, v0.z * alpha); atomicAdd(arow + 3, v0.w * alpha);
  atomicAdd(arow + 4, v1.x * alpha); atomicAdd(arow + 5, v1.y * alpha);
  atomicAdd(arow + 6, v1.z * alpha); atomicAdd(arow + 7, v1.w * alpha);
}

// head-mean + bias + BN + ELU -> h_next[N,64]
__global__ __launch_bounds__(256)
void finalize_layer(const float* __restrict__ acc, const float* __restrict__ bias,
                    const float* __restrict__ g, const float* __restrict__ bt,
                    const float* __restrict__ mu, const float* __restrict__ var,
                    float* __restrict__ hout, int nnodes) {
  long long gid = (long long)blockIdx.x * blockDim.x + threadIdx.x;
  if (gid >= (long long)nnodes * 64) return;
  int n = (int)(gid >> 6), c = (int)(gid & 63);
  const float* a = acc + (size_t)n * 256;
  float v = 0.25f * (a[c] + a[64 + c] + a[128 + c] + a[192 + c]) + bias[c];
  v = (v - mu[c]) * rsqrtf(var[c] + BN_EPS) * g[c] + bt[c];
  v = v > 0.f ? v : (__expf(v) - 1.f);                 // ELU
  hout[(size_t)n * 64 + c] = v;
}

// fc0: [N,64]@[64,32] + b, BN -> out0 (also the next layer's input)
__global__ __launch_bounds__(256)
void fc0_kernel(const float* __restrict__ h, const float* __restrict__ W,
                const float* __restrict__ b, const float* __restrict__ g,
                const float* __restrict__ bt, const float* __restrict__ mu,
                const float* __restrict__ var, float* __restrict__ out, int nnodes) {
  long long gid = (long long)blockIdx.x * blockDim.x + threadIdx.x;
  if (gid >= (long long)nnodes * 32) return;
  int n = (int)(gid >> 5), o = (int)(gid & 31);
  const float* hr = h + (size_t)n * 64;
  float v = b[o];
  #pragma unroll 8
  for (int k = 0; k < 64; ++k) v += hr[k] * W[k * 32 + o];
  v = (v - mu[o]) * rsqrtf(var[o] + BN_EPS) * g[o] + bt[o];
  out[(size_t)n * 32 + o] = v;
}

// fc1: [N,32]@[32,16] + b, ELU, BN -> out1
__global__ __launch_bounds__(256)
void fc1_kernel(const float* __restrict__ h, const float* __restrict__ W,
                const float* __restrict__ b, const float* __restrict__ g,
                const float* __restrict__ bt, const float* __restrict__ mu,
                const float* __restrict__ var, float* __restrict__ out, int nnodes) {
  long long gid = (long long)blockIdx.x * blockDim.x + threadIdx.x;
  if (gid >= (long long)nnodes * 16) return;
  int n = (int)(gid >> 4), o = (int)(gid & 15);
  const float* hr = h + (size_t)n * 32;
  float v = b[o];
  #pragma unroll
  for (int k = 0; k < 32; ++k) v += hr[k] * W[k * 16 + o];
  v = v > 0.f ? v : (__expf(v) - 1.f);                 // ELU (layers j>0)
  v = (v - mu[o]) * rsqrtf(var[o] + BN_EPS) * g[o] + bt[o];
  out[(size_t)n * 16 + o] = v;
}

extern "C" void kernel_launch(void* const* d_in, const int* in_sizes, int n_in,
                              void* d_out, int out_size, void* d_ws, size_t ws_size,
                              hipStream_t stream) {
  const int N = in_sizes[0] / 128;          // 100000
  const int E = in_sizes[1] / 2;            // 1600000
  const float*     x  = (const float*)d_in[0];
  const long long* ei = (const long long*)d_in[1];
  // params in setup_inputs() dict order
  const float* P[28];
  for (int i = 0; i < 28 && (i + 2) < n_in; ++i) P[i] = (const float*)d_in[i + 2];
  // layer i base: 8*i -> W, a_src, a_dst, b, bn_g, bn_b, bn_m, bn_v
  // fc j base: 16 + 6*j -> W, b, g, b, m, v

  float* ws = (float*)d_ws;
  size_t off = 0;
  float*    xl    = ws + off; off += (size_t)N * 256;
  float*    acc   = ws + off; off += (size_t)N * 256;
  float*    asrc  = ws + off; off += (size_t)N * 4;
  float*    adst  = ws + off; off += (size_t)N * 4;
  unsigned* mEnc  = (unsigned*)(ws + off); off += (size_t)N * 4;
  float*    denom = ws + off; off += (size_t)N * 4;
  float*    hbuf  = ws + off; off += (size_t)N * 64;
  (void)ws_size;

  float* out0 = (float*)d_out;
  float* out1 = out0 + (size_t)N * 32;

  const int T = 256;
  auto cdiv = [](long long a, long long b) { return (unsigned)((a + b - 1) / b); };
  const long long perEdge = (long long)(E + N) * 4;

  for (int L = 0; L < 2; ++L) {
    const float* Ain = L ? hbuf : x;
    const float* Wl  = P[8 * L + 0];
    const float* aS  = P[8 * L + 1];
    const float* aD  = P[8 * L + 2];
    const float* bl  = P[8 * L + 3];
    const float* bg  = P[8 * L + 4];
    const float* bb  = P[8 * L + 5];
    const float* bm  = P[8 * L + 6];
    const float* bv  = P[8 * L + 7];

    // graph-capturable memset nodes for the fp32 accumulators
    hipMemsetAsync(acc, 0, (size_t)N * 256 * sizeof(float), stream);
    hipMemsetAsync(denom, 0, (size_t)N * 4 * sizeof(float), stream);
    fill_u32<<<cdiv((long long)N * 4, T), T, 0, stream>>>(mEnc, ORD_NEG_INF, (long long)N * 4);

    if (L == 0)
      gat_gemm_wmma<128><<<cdiv(N, 32), 256, 0, stream>>>(Ain, Wl, xl, N);
    else
      gat_gemm_wmma<64><<<cdiv(N, 32), 256, 0, stream>>>(Ain, Wl, xl, N);

    alpha_kernel<<<cdiv((long long)N * 4, T), T, 0, stream>>>(xl, aS, aD, asrc, adst, N);
    edge_max<<<cdiv(perEdge, T), T, 0, stream>>>(ei, asrc, adst, mEnc, E, N);
    edge_sum<<<cdiv(perEdge, T), T, 0, stream>>>(ei, asrc, adst, mEnc, denom, E, N);
    edge_msg<<<cdiv((long long)(E + N), 8), T, 0, stream>>>(ei, asrc, adst, mEnc, denom,
                                                            xl, acc, E, N);
    finalize_layer<<<cdiv((long long)N * 64, T), T, 0, stream>>>(acc, bl, bg, bb, bm, bv,
                                                                 hbuf, N);
  }

  fc0_kernel<<<cdiv((long long)N * 32, T), T, 0, stream>>>(
      hbuf, P[16], P[17], P[18], P[19], P[20], P[21], out0, N);
  fc1_kernel<<<cdiv((long long)N * 16, T), T, 0, stream>>>(
      out0, P[22], P[23], P[24], P[25], P[26], P[27], out1, N);
}
